// KiperwasserDependencyParser_28544352649427
// MI455X (gfx1250) — matrix operand order repaired and checked by reading, compiled, not run on hardware
//
#include <hip/hip_runtime.h>

// ---------------- types ----------------
typedef _Float16 h8   __attribute__((ext_vector_type(8)));
typedef _Float16 v16h __attribute__((ext_vector_type(16)));
typedef float    v8f  __attribute__((ext_vector_type(8)));
typedef unsigned int u32x4 __attribute__((ext_vector_type(4)));
typedef int          i32x4 __attribute__((ext_vector_type(4)));
typedef int          i32x8 __attribute__((ext_vector_type(8)));

// ---------------- problem constants ----------------
#define TSEQ 512
#define HID  512
#define WD   1024
#define TD   64
#define IN0  1088          // WD + TD
#define IN1  1024          // 2*HID
#define G4   2048          // 4*HID
#define MLPH 100
#define SHLD 128           // padded ld for sh/sm

// ---------------- GEMM tiling ----------------
#define BM 128
#define BN 128
#define BK 32
#define LDT 40             // LDS tile ld in halfs (64B row + 16B pad)

#if __has_builtin(__builtin_amdgcn_tensor_load_to_lds)
#define HAVE_TDM 1
#endif

__device__ __forceinline__ float sigmoidf_(float x) {
    return 1.0f / (1.0f + __expf(-x));
}

#ifdef HAVE_TDM
// ---------------------------------------------------------------------------
// Tensor Data Mover: load a tile_rows x 32(f16) tile from a row-major tensor
// (row stride = stride_elems) into LDS at lds_off, padding each 64B row with
// 16B so the LDS pitch is LDT=40 halfs. rows/cols_rem give the remaining
// tensor extent from the tile origin -> OOB rows are zero-filled by TDM.
// D# layout per CDNA5 ISA 8.3/8.4 (group0 128b, group1 256b).
// ---------------------------------------------------------------------------
__device__ __forceinline__ void tdm_load_tile(
    const _Float16* g, unsigned lds_off, unsigned rows_rem,
    unsigned cols_rem, unsigned stride_elems, unsigned tile_rows)
{
    unsigned long long ga = (unsigned long long)(uintptr_t)g;
    u32x4 g0;
    g0[0] = 1u;                                            // count=1, user D#
    g0[1] = lds_off;                                       // lds_addr (bytes)
    g0[2] = (unsigned)ga;                                  // global_addr[31:0]
    g0[3] = (unsigned)((ga >> 32) & 0x01ffffffu)           // global_addr[56:32]
          | (2u << 30);                                    // type = 2 ("image")

    i32x8 g1;
    // data_size=1 (2B), pad_enable=1, pad_interval=3 (16 DW), pad_amount=3 (4 DW)
    g1[0] = (int)((1u << 16) | (1u << 20) | (3u << 22) | (3u << 25));
    g1[1] = (int)((cols_rem & 0xffffu) << 16);             // tensor_dim0[15:0]
    g1[2] = (int)((cols_rem >> 16) | ((rows_rem & 0xffffu) << 16));
    g1[3] = (int)((rows_rem >> 16) | ((unsigned)BK << 16)); // tile_dim0 = 32
    g1[4] = (int)(tile_rows & 0xffffu);                    // tile_dim1; tile_dim2=0
    g1[5] = (int)stride_elems;                             // tensor_dim0_stride lo
    g1[6] = 0;                                             // stride hi, dim1_stride lo
    g1[7] = 0;
    i32x4 z4 = {0, 0, 0, 0};
#if __clang_major__ >= 23
    i32x8 z8 = {0, 0, 0, 0, 0, 0, 0, 0};
    __builtin_amdgcn_tensor_load_to_lds(g0, g1, z4, z4, z8, 0);
#else
    __builtin_amdgcn_tensor_load_to_lds(g0, g1, z4, z4, 0);
#endif
}
#endif

// A fragment: 16x32 f16, rows M in lane&15, K split across lane halves per ISA table.
__device__ __forceinline__ v16h load_a_frag(const _Float16* tile, int row0, int ld) {
    int lane = threadIdx.x & 31;
    int m = lane & 15, hi = lane >> 4;
    v16h a;
#pragma unroll
    for (int v = 0; v < 8; ++v) {
        int kb = (v < 4) ? (hi * 8 + v * 2) : (16 + hi * 8 + (v - 4) * 2);
        a[2 * v]     = tile[(row0 + m) * ld + kb];
        a[2 * v + 1] = tile[(row0 + m) * ld + kb + 1];
    }
    return a;
}

// B fragment: 32x16 f16 (K x N). n = lane&15; k = (lane>>4)*8 + (e&7) + (e>>3)*16.
__device__ __forceinline__ v16h load_b_frag(const _Float16* tile, int n0, int ld) {
    int lane = threadIdx.x & 31;
    int n = lane & 15, hi = lane >> 4;
    v16h b;
#pragma unroll
    for (int e = 0; e < 16; ++e)
        b[e] = tile[(n0 + n) * ld + hi * 8 + (e & 7) + (e >> 3) * 16];
    return b;
}

// ---------------------------------------------------------------------------
// C[M,N] (f32) = A[M,K] (f16, row-major) * B[N,K]^T (f16, row-major) + bias[N]
// K % 32 == 0. 256 threads = 8 waves, block tile 128x128, wave tile 32x64.
// TDM path: double-buffered tensor_load_to_lds (wave 0 issues, TENSORcnt
// pipelining) so HBM/L2 traffic overlaps the v_wmma stream.
// ---------------------------------------------------------------------------
__global__ __launch_bounds__(256) void wmma_gemm_nt(
    const _Float16* __restrict__ A, const _Float16* __restrict__ B,
    const float* __restrict__ bias, float* __restrict__ C,
    int M, int N, int K, int ldc)
{
    __shared__ __align__(128) _Float16 As[2][BM * LDT];
    __shared__ __align__(128) _Float16 Bs[2][BN * LDT];

    const int tid  = threadIdx.x;
    const int wave = tid >> 5;
    const int bm = blockIdx.y * BM;
    const int bn = blockIdx.x * BN;
    const int wm = (wave >> 1) * 32;   // 0,32,64,96
    const int wn = (wave & 1) * 64;    // 0,64

    v8f acc[2][4];
#pragma unroll
    for (int mi = 0; mi < 2; ++mi)
#pragma unroll
        for (int ni = 0; ni < 4; ++ni) {
            v8f z = {0.f, 0.f, 0.f, 0.f, 0.f, 0.f, 0.f, 0.f};
            acc[mi][ni] = z;
        }

    const int nk = K / BK;

#ifdef HAVE_TDM
    const bool issuer = (tid < 32);    // wave 0 drives the TDM
    const unsigned offA[2] = { (unsigned)(uintptr_t)&As[0][0],
                               (unsigned)(uintptr_t)&As[1][0] };
    const unsigned offB[2] = { (unsigned)(uintptr_t)&Bs[0][0],
                               (unsigned)(uintptr_t)&Bs[1][0] };
    if (issuer) {
        tdm_load_tile(A + (size_t)bm * K, offA[0], (unsigned)(M - bm),
                      (unsigned)K, (unsigned)K, BM);
        tdm_load_tile(B + (size_t)bn * K, offB[0], (unsigned)(N - bn),
                      (unsigned)K, (unsigned)K, BN);
    }
    for (int ki = 0; ki < nk; ++ki) {
        const int cur = ki & 1, nxt = cur ^ 1;
        const int k0 = ki * BK;
        if (issuer) {
            if (ki + 1 < nk) {
                // prefetch next K-slab into the other buffer, then wait until
                // only those 2 new ops are outstanding (current pair landed).
                tdm_load_tile(A + (size_t)bm * K + k0 + BK, offA[nxt],
                              (unsigned)(M - bm), (unsigned)(K - k0 - BK),
                              (unsigned)K, BM);
                tdm_load_tile(B + (size_t)bn * K + k0 + BK, offB[nxt],
                              (unsigned)(N - bn), (unsigned)(K - k0 - BK),
                              (unsigned)K, BN);
                __builtin_amdgcn_s_wait_tensorcnt(2);
            } else {
                __builtin_amdgcn_s_wait_tensorcnt(0);
            }
        }
        __syncthreads();

        v16h bf[4];
#pragma unroll
        for (int ni = 0; ni < 4; ++ni)
            bf[ni] = load_b_frag(&Bs[cur][0], wn + ni * 16, LDT);
#pragma unroll
        for (int mi = 0; mi < 2; ++mi) {
            v16h af = load_a_frag(&As[cur][0], wm + mi * 16, LDT);
#pragma unroll
            for (int ni = 0; ni < 4; ++ni) {
                acc[mi][ni] = __builtin_amdgcn_wmma_f32_16x16x32_f16(
                    false, af, false, bf[ni], (short)0, acc[mi][ni], false, false);
            }
        }
        __syncthreads();
    }
#else
    // fallback: manual staged copy (round-1 path)
    for (int ki = 0; ki < nk; ++ki) {
        const int k0 = ki * BK;
#pragma unroll
        for (int i = tid; i < (BM * BK) / 8; i += 256) {
            int r = i >> 2, c8 = (i & 3) * 8;
            h8 v = {0, 0, 0, 0, 0, 0, 0, 0};
            int gr = bm + r;
            if (gr < M) v = *(const h8*)(A + (size_t)gr * K + k0 + c8);
            *(h8*)(&As[0][0] + r * LDT + c8) = v;
        }
#pragma unroll
        for (int i = tid; i < (BN * BK) / 8; i += 256) {
            int r = i >> 2, c8 = (i & 3) * 8;
            h8 v = {0, 0, 0, 0, 0, 0, 0, 0};
            int gr = bn + r;
            if (gr < N) v = *(const h8*)(B + (size_t)gr * K + k0 + c8);
            *(h8*)(&Bs[0][0] + r * LDT + c8) = v;
        }
        __syncthreads();
        v16h bf[4];
#pragma unroll
        for (int ni = 0; ni < 4; ++ni)
            bf[ni] = load_b_frag(&Bs[0][0], wn + ni * 16, LDT);
#pragma unroll
        for (int mi = 0; mi < 2; ++mi) {
            v16h af = load_a_frag(&As[0][0], wm + mi * 16, LDT);
#pragma unroll
            for (int ni = 0; ni < 4; ++ni) {
                acc[mi][ni] = __builtin_amdgcn_wmma_f32_16x16x32_f16(
                    false, af, false, bf[ni], (short)0, acc[mi][ni], false, false);
            }
        }
        __syncthreads();
    }
#endif

    // epilogue: C/D layout — n = lane&15, m = g + (lane>>4)*8
    const int lane = tid & 31;
    const int ln = lane & 15, lh = lane >> 4;
#pragma unroll
    for (int mi = 0; mi < 2; ++mi)
#pragma unroll
        for (int ni = 0; ni < 4; ++ni)
#pragma unroll
            for (int g = 0; g < 8; ++g) {
                int m = bm + wm + mi * 16 + lh * 8 + g;
                int n = bn + wn + ni * 16 + ln;
                if (m < M && n < N) {
                    float b = bias ? bias[n] : 0.0f;
                    C[(size_t)m * ldc + n] = acc[mi][ni][g] + b;
                }
            }
}

// ---------------------------------------------------------------------------
// Strided f32 -> f16 conversion (also handles W1 column halves)
// ---------------------------------------------------------------------------
__global__ void cvt_f32_f16(const float* __restrict__ src, _Float16* __restrict__ dst,
                            int rows, int cols, int sld, int dld)
{
    int total = rows * cols;
    for (int idx = blockIdx.x * blockDim.x + threadIdx.x; idx < total;
         idx += gridDim.x * blockDim.x) {
        int r = idx / cols, c = idx - r * cols;
        dst[(size_t)r * dld + c] = (_Float16)src[(size_t)r * sld + c];
    }
}

// ---------------------------------------------------------------------------
// Embedding gather + concat -> X0 f16 [T][IN0]
// ---------------------------------------------------------------------------
__global__ void embed_kernel(const int* __restrict__ widx, const int* __restrict__ pidx,
                             const float* __restrict__ wemb, const float* __restrict__ pemb,
                             _Float16* __restrict__ x0)
{
    int t = blockIdx.x;
    int wi = widx[t], pi = pidx[t];
    for (int d = threadIdx.x; d < WD; d += blockDim.x)
        x0[(size_t)t * IN0 + d] = (_Float16)wemb[(size_t)wi * WD + d];
    for (int d = threadIdx.x; d < TD; d += blockDim.x)
        x0[(size_t)t * IN0 + WD + d] = (_Float16)pemb[(size_t)pi * TD + d];
}

// ---------------------------------------------------------------------------
// LSTM recurrent scan: one block per direction (grid=2), 1024 threads.
// z[j] = xw[t][j] + sum_k h[k]*whh[j][k]; PyTorch gate order i,f,g,o.
// Writes h into the concat buffer: out[t][dir*HID + j].
// ---------------------------------------------------------------------------
__global__ __launch_bounds__(1024) void lstm_scan(
    const float* __restrict__ xw_f, const float* __restrict__ xw_b,
    const _Float16* __restrict__ whh_f, const _Float16* __restrict__ whh_b,
    float* __restrict__ out, int out_ld)
{
    __shared__ float hbuf[HID];
    __shared__ float zbuf[G4];

    const int dir = blockIdx.x;                 // 0 = fwd, 1 = bwd
    const float*    xw  = dir ? xw_b  : xw_f;
    const _Float16* whh = dir ? whh_b : whh_f;
    const int out_off = dir * HID;
    const int tid = threadIdx.x;

    if (tid < HID) hbuf[tid] = 0.0f;
    float c = 0.0f;
    __syncthreads();

    for (int t = 0; t < TSEQ; ++t) {
        const int tt = dir ? (TSEQ - 1 - t) : t;
        for (int j = tid; j < G4; j += 1024) {
            float acc = xw[(size_t)tt * G4 + j];
            const h8* wr = (const h8*)(whh + (size_t)j * HID);
#pragma unroll 4
            for (int k8 = 0; k8 < HID / 8; ++k8) {
                h8 w = wr[k8];
#pragma unroll
                for (int e = 0; e < 8; ++e)
                    acc += (float)w[e] * hbuf[k8 * 8 + e];
            }
            zbuf[j] = acc;
        }
        __syncthreads();
        if (tid < HID) {
            float zi = zbuf[tid];
            float zf = zbuf[tid + HID];
            float zg = zbuf[tid + 2 * HID];
            float zo = zbuf[tid + 3 * HID];
            c = sigmoidf_(zf) * c + sigmoidf_(zi) * tanhf(zg);
            float hn = sigmoidf_(zo) * tanhf(c);
            hbuf[tid] = hn;
            out[(size_t)tt * out_ld + out_off + tid] = hn;
        }
        __syncthreads();
    }
}

// ---------------------------------------------------------------------------
// scores + column softmax: block = modifier column md; threads cover heads.
// ---------------------------------------------------------------------------
__global__ __launch_bounds__(256) void score_softmax(
    const float* __restrict__ sh, const float* __restrict__ sm,
    const float* __restrict__ b1, const float* __restrict__ W2,
    const float* __restrict__ b2, float* __restrict__ out)
{
    __shared__ float smv[MLPH];
    __shared__ float w2s[MLPH];
    __shared__ float sval[TSEQ];
    __shared__ float red[256];

    const int md = blockIdx.x;
    const int tid = threadIdx.x;

    if (tid < MLPH) {
        smv[tid] = sm[(size_t)md * SHLD + tid] + b1[tid];
        w2s[tid] = W2[tid];
    }
    __syncthreads();

    const float bias2 = b2[0];
#pragma unroll
    for (int r = 0; r < 2; ++r) {
        int hd = tid + r * 256;
        const float* shr = sh + (size_t)hd * SHLD;
        float acc = bias2;
        for (int k = 0; k < MLPH; ++k)
            acc += w2s[k] * tanhf(shr[k] + smv[k]);
        sval[hd] = acc;
    }
    __syncthreads();

    float m = fmaxf(sval[tid], sval[tid + 256]);
    red[tid] = m;
    __syncthreads();
    for (int s = 128; s > 0; s >>= 1) {
        if (tid < s) red[tid] = fmaxf(red[tid], red[tid + s]);
        __syncthreads();
    }
    float mx = red[0];
    __syncthreads();

    float e0 = __expf(sval[tid] - mx);
    float e1 = __expf(sval[tid + 256] - mx);
    red[tid] = e0 + e1;
    __syncthreads();
    for (int s = 128; s > 0; s >>= 1) {
        if (tid < s) red[tid] += red[tid + s];
        __syncthreads();
    }
    float inv = 1.0f / red[0];

    out[(size_t)tid * TSEQ + md] = e0 * inv;
    out[(size_t)(tid + 256) * TSEQ + md] = e1 * inv;
}

// ---------------------------------------------------------------------------
// host driver
// ---------------------------------------------------------------------------
extern "C" void kernel_launch(void* const* d_in, const int* in_sizes, int n_in,
                              void* d_out, int out_size, void* d_ws, size_t ws_size,
                              hipStream_t stream)
{
    const int*   word_idx = (const int*)d_in[0];
    const int*   pos_idx  = (const int*)d_in[1];
    const float* word_emb = (const float*)d_in[2];
    const float* pos_emb  = (const float*)d_in[3];
    const float* w_ih_l0f = (const float*)d_in[4];
    const float* w_hh_l0f = (const float*)d_in[5];
    const float* b_l0f    = (const float*)d_in[6];
    const float* w_ih_l0b = (const float*)d_in[7];
    const float* w_hh_l0b = (const float*)d_in[8];
    const float* b_l0b    = (const float*)d_in[9];
    const float* w_ih_l1f = (const float*)d_in[10];
    const float* w_hh_l1f = (const float*)d_in[11];
    const float* b_l1f    = (const float*)d_in[12];
    const float* w_ih_l1b = (const float*)d_in[13];
    const float* w_hh_l1b = (const float*)d_in[14];
    const float* b_l1b    = (const float*)d_in[15];
    const float* W1       = (const float*)d_in[16];
    const float* b1       = (const float*)d_in[17];
    const float* W2       = (const float*)d_in[18];
    const float* b2       = (const float*)d_in[19];
    float* out = (float*)d_out;

    char* ws = (char*)d_ws;
    size_t off = 0;
    auto carve = [&](size_t bytes) -> void* {
        void* p = ws + off;
        off += (bytes + 255) & ~(size_t)255;
        return p;
    };

    _Float16* x0h     = (_Float16*)carve((size_t)TSEQ * IN0 * 2);
    _Float16* wih0f_h = (_Float16*)carve((size_t)G4 * IN0 * 2);
    _Float16* wih0b_h = (_Float16*)carve((size_t)G4 * IN0 * 2);
    _Float16* wih1f_h = (_Float16*)carve((size_t)G4 * IN1 * 2);
    _Float16* wih1b_h = (_Float16*)carve((size_t)G4 * IN1 * 2);
    _Float16* whh0f_h = (_Float16*)carve((size_t)G4 * HID * 2);
    _Float16* whh0b_h = (_Float16*)carve((size_t)G4 * HID * 2);
    _Float16* whh1f_h = (_Float16*)carve((size_t)G4 * HID * 2);
    _Float16* whh1b_h = (_Float16*)carve((size_t)G4 * HID * 2);
    _Float16* w1a_h   = (_Float16*)carve((size_t)MLPH * IN1 * 2);
    _Float16* w1b_h   = (_Float16*)carve((size_t)MLPH * IN1 * 2);
    float* xw0f = (float*)carve((size_t)TSEQ * G4 * 4);
    float* xw0b = (float*)carve((size_t)TSEQ * G4 * 4);
    float* xw1f = (float*)carve((size_t)TSEQ * G4 * 4);
    float* xw1b = (float*)carve((size_t)TSEQ * G4 * 4);
    float* h0   = (float*)carve((size_t)TSEQ * IN1 * 4);
    float* h1   = (float*)carve((size_t)TSEQ * IN1 * 4);
    _Float16* h0h = (_Float16*)carve((size_t)TSEQ * IN1 * 2);
    _Float16* h1h = (_Float16*)carve((size_t)TSEQ * IN1 * 2);
    float* sh = (float*)carve((size_t)TSEQ * SHLD * 4);
    float* sm = (float*)carve((size_t)TSEQ * SHLD * 4);
    (void)ws_size; (void)in_sizes; (void)n_in; (void)out_size;

    auto cvt = [&](const float* s, _Float16* d, int r, int c, int sld, int dld) {
        int total = r * c;
        int blocks = (total + 255) / 256;
        if (blocks > 2048) blocks = 2048;
        cvt_f32_f16<<<dim3(blocks), dim3(256), 0, stream>>>(s, d, r, c, sld, dld);
    };
    auto gemm = [&](const _Float16* A, const _Float16* B, const float* bias,
                    float* C, int M, int N, int K, int ldc) {
        dim3 grid((N + BN - 1) / BN, (M + BM - 1) / BM);
        wmma_gemm_nt<<<grid, dim3(256), 0, stream>>>(A, B, bias, C, M, N, K, ldc);
    };

    // --- weight conversions (f32 -> f16) ---
    cvt(w_ih_l0f, wih0f_h, G4, IN0, IN0, IN0);
    cvt(w_ih_l0b, wih0b_h, G4, IN0, IN0, IN0);
    cvt(w_ih_l1f, wih1f_h, G4, IN1, IN1, IN1);
    cvt(w_ih_l1b, wih1b_h, G4, IN1, IN1, IN1);
    cvt(w_hh_l0f, whh0f_h, G4, HID, HID, HID);
    cvt(w_hh_l0b, whh0b_h, G4, HID, HID, HID);
    cvt(w_hh_l1f, whh1f_h, G4, HID, HID, HID);
    cvt(w_hh_l1b, whh1b_h, G4, HID, HID, HID);
    cvt(W1,        w1a_h, MLPH, IN1, 2 * IN1, IN1);   // W1[:, :1024]
    cvt(W1 + IN1,  w1b_h, MLPH, IN1, 2 * IN1, IN1);   // W1[:, 1024:]

    // --- embeddings ---
    embed_kernel<<<dim3(TSEQ), dim3(256), 0, stream>>>(word_idx, pos_idx,
                                                       word_emb, pos_emb, x0h);

    // --- layer 0: input projections (WMMA + TDM) then recurrent scan ---
    gemm(x0h, wih0f_h, b_l0f, xw0f, TSEQ, G4, IN0, G4);
    gemm(x0h, wih0b_h, b_l0b, xw0b, TSEQ, G4, IN0, G4);
    lstm_scan<<<dim3(2), dim3(1024), 0, stream>>>(xw0f, xw0b, whh0f_h, whh0b_h,
                                                  h0, IN1);
    cvt(h0, h0h, TSEQ, IN1, IN1, IN1);

    // --- layer 1 ---
    gemm(h0h, wih1f_h, b_l1f, xw1f, TSEQ, G4, IN1, G4);
    gemm(h0h, wih1b_h, b_l1b, xw1b, TSEQ, G4, IN1, G4);
    lstm_scan<<<dim3(2), dim3(1024), 0, stream>>>(xw1f, xw1b, whh1f_h, whh1b_h,
                                                  h1, IN1);
    cvt(h1, h1h, TSEQ, IN1, IN1, IN1);

    // --- edge-scorer MLP projections (WMMA, N=100 inside one 128 tile,
    //     TDM zero-fills the OOB rows) ---
    gemm(h1h, w1a_h, nullptr, sh, TSEQ, MLPH, IN1, SHLD);
    gemm(h1h, w1b_h, nullptr, sm, TSEQ, MLPH, IN1, SHLD);

    // --- pairwise scores + softmax over heads ---
    score_softmax<<<dim3(TSEQ), dim3(256), 0, stream>>>(sh, sm, b1, W2, b2, out);
}
